// GSVAN_89309549953297
// MI455X (gfx1250) — compile-verified, matrix operand order checked by reading, not run on hardware
//
#include <hip/hip_runtime.h>
#include <stdint.h>

// ---------------- problem constants ----------------
constexpr int kS = 16;      // species / steps
constexpr int kC = 64;      // categories
constexpr int kH = 512;     // hidden
constexpr int kP = 12;      // posenc width (2*NF)
constexpr int kB = 32768;   // batch

typedef __attribute__((ext_vector_type(16))) _Float16 v16h;
typedef __attribute__((ext_vector_type(8)))  _Float16 v8h;
typedef __attribute__((ext_vector_type(8)))  float    v8f;
typedef uint32_t u32x4 __attribute__((ext_vector_type(4)));
typedef uint32_t u32x8 __attribute__((ext_vector_type(8)));

union FragU { v16h v; v8h h[2]; };

// base must point at (row start + khalf*8); kb is the K-chunk base (multiple of 32)
__device__ __forceinline__ v16h load_frag(const _Float16* __restrict__ base, int kb) {
    FragU f;
    f.h[0] = *(const v8h*)(base + kb);
    f.h[1] = *(const v8h*)(base + kb + 16);
    return f.v;
}

__device__ __forceinline__ v8f wmma_f16(v16h a, v16h b, v8f c) {
    return __builtin_amdgcn_wmma_f32_16x16x32_f16(
        /*neg_a=*/false, a, /*neg_b=*/false, b,
        /*c_mod=*/(short)0, c, /*reuse_a=*/false, /*reuse_b=*/false);
}

__device__ __forceinline__ float fsigmoid(float x) { return 1.0f / (1.0f + __expf(-x)); }
__device__ __forceinline__ float fsilu(float x)    { return x / (1.0f + __expf(-x)); }

// ---------------- utility kernels ----------------
__global__ void cvt_f32_to_f16(const float* __restrict__ in, _Float16* __restrict__ out, int n) {
    int i = blockIdx.x * blockDim.x + threadIdx.x;
    if (i < n) out[i] = (_Float16)in[i];
}

__global__ void zero_f32(float* __restrict__ p, int n) {
    int i = blockIdx.x * blockDim.x + threadIdx.x;
    if (i < n) p[i] = 0.0f;
}

// pe[b, 0:6] = sin(2^k*pi*t), pe[b, 6:12] = cos(2^k*pi*t)
__global__ void posenc_kernel(const float* __restrict__ t, float* __restrict__ pe) {
    int b = blockIdx.x * blockDim.x + threadIdx.x;
    if (b >= kB) return;
    float tv = t[b];
    float f = 3.14159265358979323846f;
#pragma unroll
    for (int k = 0; k < 6; ++k) {
        float ang = tv * f;
        pe[b * kP + k]     = __sinf(ang);
        pe[b * kP + 6 + k] = __cosf(ang);
        f *= 2.0f;
    }
}

// x1 = silu(pe @ te_W1.T + te_b1) stored as f16  [B,H]
__global__ void te1_kernel(const float* __restrict__ pe, const float* __restrict__ W1,
                           const float* __restrict__ b1, _Float16* __restrict__ x1) {
    int idx = blockIdx.x * blockDim.x + threadIdx.x;  // b*H + j
    int b = idx >> 9, j = idx & (kH - 1);
    float acc = b1[j];
#pragma unroll
    for (int k = 0; k < kP; ++k) acc += pe[b * kP + k] * W1[j * kP + k];
    x1[idx] = (_Float16)fsilu(acc);
}

// h0 = x1 @ te_W2.T + te_b2 ; writes c0 (f32) and h0 (f16).  WMMA GEMM.
__global__ __launch_bounds__(256) void te2_kernel(const _Float16* __restrict__ x1,
                                                  const _Float16* __restrict__ w16,
                                                  const float* __restrict__ b2,
                                                  float* __restrict__ cbuf,
                                                  _Float16* __restrict__ hout) {
    int wave = threadIdx.x >> 5, lane = threadIdx.x & 31;
    int wid = blockIdx.x * 8 + wave;            // 65536 wave tiles
    int tile_m = wid >> 5, tile_n = wid & 31;   // 2048 x 32
    int lrow = lane & 15, khalf = lane >> 4;
    int Mbase = tile_m * 16;
    int col = tile_n * 16 + lrow;

    const _Float16* abase = x1 + (size_t)(Mbase + lrow) * kH + khalf * 8;
    const _Float16* bbase = w16 + (size_t)col * kH + khalf * 8;
    v8f acc = {};
#pragma unroll 4
    for (int kb = 0; kb < kH; kb += 32) {
        acc = wmma_f16(load_frag(abase, kb), load_frag(bbase, kb), acc);
    }
    float bias = b2[col];
#pragma unroll
    for (int r = 0; r < 8; ++r) {
        int row = Mbase + r + 8 * khalf;
        float h0 = acc[r] + bias;
        size_t idx = (size_t)row * kH + col;
        cbuf[idx] = h0;
        hout[idx] = (_Float16)h0;
    }
}

// species operator: so[c, n] = silu(c*s1W + s1b) @ s2W.T + s2b   -> f16 [C,H]
__global__ void species_kernel(const float* __restrict__ s1W, const float* __restrict__ s1b,
                               const float* __restrict__ s2W, const float* __restrict__ s2b,
                               _Float16* __restrict__ so16) {
    __shared__ float x[kH];
    int c = blockIdx.x;
    for (int j = threadIdx.x; j < kH; j += blockDim.x)
        x[j] = fsilu((float)c * s1W[j] + s1b[j]);
    __syncthreads();
    for (int n = threadIdx.x; n < kH; n += blockDim.x) {
        const float* wrow = s2W + (size_t)n * kH;
        float acc = s2b[n];
        for (int j = 0; j < kH; ++j) acc += x[j] * wrow[j];
        so16[(size_t)c * kH + n] = (_Float16)acc;
    }
}

// Fused: gates = (pe+cond) @ W_ih.T + h @ W_hh.T + b_ih + b_hh ; LSTM update.
// Each wave: 32(M) x 16(N hidden) tile, 4 gates -> 8 accumulators.
__global__ __launch_bounds__(256) void lstm_kernel(const _Float16* __restrict__ hin,
                                                   const _Float16* __restrict__ whh16,
                                                   const float* __restrict__ pe,
                                                   const float* __restrict__ cond,
                                                   const float* __restrict__ W_ih,
                                                   const float* __restrict__ b_ih,
                                                   const float* __restrict__ b_hh,
                                                   float* __restrict__ cbuf,
                                                   _Float16* __restrict__ hout) {
    int wave = threadIdx.x >> 5, lane = threadIdx.x & 31;
    int wid = blockIdx.x * 8 + wave;            // 32768 wave tiles
    int tile_m = wid >> 5, tile_n = wid & 31;   // 1024 x 32
    int lrow = lane & 15, khalf = lane >> 4;
    int Mbase = tile_m * 32;
    int col = tile_n * 16 + lrow;               // hidden unit 0..511

    const _Float16* ab0 = hin + (size_t)(Mbase + lrow) * kH + khalf * 8;
    const _Float16* ab1 = ab0 + (size_t)16 * kH;
    const _Float16* bb[4];
#pragma unroll
    for (int g = 0; g < 4; ++g)
        bb[g] = whh16 + (size_t)(col + g * kH) * kH + khalf * 8;

    v8f acc[4][2] = {{{}, {}}, {{}, {}}, {{}, {}}, {{}, {}}};
#pragma unroll 2
    for (int kb = 0; kb < kH; kb += 32) {
        v16h a0 = load_frag(ab0, kb);
        v16h a1 = load_frag(ab1, kb);
#pragma unroll
        for (int g = 0; g < 4; ++g) {
            v16h b = load_frag(bb[g], kb);
            acc[g][0] = wmma_f16(a0, b, acc[g][0]);
            acc[g][1] = wmma_f16(a1, b, acc[g][1]);
        }
    }

    // input-projection (K=12) + biases, per gate (col fixed per lane)
    float wih[4][kP], bias[4];
#pragma unroll
    for (int g = 0; g < 4; ++g) {
        int cg = col + g * kH;
        bias[g] = b_ih[cg] + b_hh[cg];
#pragma unroll
        for (int k = 0; k < kP; ++k) wih[g][k] = W_ih[cg * kP + k];
    }
#pragma unroll
    for (int m = 0; m < 2; ++m) {
#pragma unroll
        for (int r = 0; r < 8; ++r) {
            int row = Mbase + m * 16 + r + 8 * khalf;
            float x[kP];
#pragma unroll
            for (int k = 0; k < kP; ++k) x[k] = pe[row * kP + k] + cond[row * kP + k];
            float si = bias[0], sf = bias[1], sg = bias[2], so = bias[3];
#pragma unroll
            for (int k = 0; k < kP; ++k) {
                si += x[k] * wih[0][k]; sf += x[k] * wih[1][k];
                sg += x[k] * wih[2][k]; so += x[k] * wih[3][k];
            }
            float iv = fsigmoid(acc[0][m][r] + si);
            float fv = fsigmoid(acc[1][m][r] + sf);
            float gv = tanhf(acc[2][m][r] + sg);
            float ov = fsigmoid(acc[3][m][r] + so);
            size_t idx = (size_t)row * kH + col;
            float cn = fv * cbuf[idx] + iv * gv;
            cbuf[idx] = cn;
            hout[idx] = (_Float16)(ov * tanhf(cn));
        }
    }
}

// logits = h @ so.T : [B,64].  so (64x512 f16) staged into LDS via the Tensor
// Data Mover (tensor_load_to_lds) with row padding (520-half LDS stride) to
// break bank conflicts.  One wave per 32-row tile, 4 N-tiles -> 8 accumulators.
// B fragments are pulled from LDS with explicit ds_load_b128 inline asm (the
// generic-pointer path was lowering to flat); one s_wait_dscnt per K-chunk.
constexpr int LSTRIDE = kH + 8;                     // 520 halfs = 1040 B per row
__global__ __launch_bounds__(256) void logits_kernel(const _Float16* __restrict__ hin,
                                                     const _Float16* __restrict__ so16,
                                                     float* __restrict__ logits) {
    __shared__ __attribute__((aligned(16))) _Float16 sB[kC * LSTRIDE];

    if (threadIdx.x < 32) {                         // wave 0 issues the TDM DMA
        uint64_t ga = (uint64_t)(uintptr_t)so16;
        u32x4 g0;
        g0[0] = 1u;                                 // count=1 (valid user D#)
        g0[1] = 0u;                                 // lds_addr = 0 (sB at LDS base)
        g0[2] = (uint32_t)(ga & 0xffffffffu);       // global_addr[31:0]
        g0[3] = (uint32_t)((ga >> 32) & 0x01ffffffu) | (2u << 30);  // addr[56:32] | type=2
        u32x8 g1;
        g1[0] = (1u << 16)                          // data_size = 2 bytes
              | (1u << 20)                          // pad_enable
              | (7u << 22)                          // pad_interval = 256 dwords (one row)
              | (3u << 25);                         // pad_amount = 4 dwords (16 B)
        g1[1] = ((uint32_t)kH & 0xffffu) << 16;     // tensor_dim0[15:0] = 512
        g1[2] = ((uint32_t)kC & 0xffffu) << 16;     // tensor_dim1[15:0] = 64
        g1[3] = ((uint32_t)kH & 0xffffu) << 16;     // tile_dim0 = 512
        g1[4] = (uint32_t)kC;                       // tile_dim1 = 64, tile_dim2 = 0
        g1[5] = (uint32_t)kH;                       // tensor_dim0_stride = 512
        g1[6] = 0u;
        g1[7] = 0u;
        u32x4 gz = {0u, 0u, 0u, 0u};                // groups 2/3 unused (2-D tile)
        asm volatile("tensor_load_to_lds %0, %1, %2, %3"
                     :: "s"(g0), "s"(g1), "s"(gz), "s"(gz) : "memory");
        __builtin_amdgcn_s_wait_tensorcnt(0);
    }
    __syncthreads();

    int wave = threadIdx.x >> 5, lane = threadIdx.x & 31;
    int tile_m = blockIdx.x * 8 + wave;             // 1024 tiles of 32 rows
    int lrow = lane & 15, khalf = lane >> 4;
    int Mbase = tile_m * 32;

    const _Float16* ab0 = hin + (size_t)(Mbase + lrow) * kH + khalf * 8;
    const _Float16* ab1 = ab0 + (size_t)16 * kH;
    // 32-bit LDS byte address (generic addr low bits == LDS offset)
    uint32_t baddr = (uint32_t)(uintptr_t)(sB + lrow * LSTRIDE + khalf * 8);

    v8f acc[4][2] = {{{}, {}}, {{}, {}}, {{}, {}}, {{}, {}}};
#pragma unroll 2
    for (int kb = 0; kb < kH; kb += 32) {
        v16h a0 = load_frag(ab0, kb);
        v16h a1 = load_frag(ab1, kb);
        FragU f0, f1, f2, f3;
        // 4 B fragments (one per 16-col N tile, N-tile stride = 16*LSTRIDE = 8320
        // halfs -> byte offsets 0, 16640, 33280, 49920; +32 B for the high K half).
        asm volatile(
            "ds_load_b128 %0, %8\n\t"
            "ds_load_b128 %1, %8 offset:32\n\t"
            "ds_load_b128 %2, %8 offset:16640\n\t"
            "ds_load_b128 %3, %8 offset:16672\n\t"
            "ds_load_b128 %4, %8 offset:33280\n\t"
            "ds_load_b128 %5, %8 offset:33312\n\t"
            "ds_load_b128 %6, %8 offset:49920\n\t"
            "ds_load_b128 %7, %8 offset:49952\n\t"
            "s_wait_dscnt 0x0"
            : "=v"(f0.h[0]), "=v"(f0.h[1]), "=v"(f1.h[0]), "=v"(f1.h[1]),
              "=v"(f2.h[0]), "=v"(f2.h[1]), "=v"(f3.h[0]), "=v"(f3.h[1])
            : "v"(baddr + (uint32_t)(kb * 2)));
        acc[0][0] = wmma_f16(a0, f0.v, acc[0][0]);
        acc[0][1] = wmma_f16(a1, f0.v, acc[0][1]);
        acc[1][0] = wmma_f16(a0, f1.v, acc[1][0]);
        acc[1][1] = wmma_f16(a1, f1.v, acc[1][1]);
        acc[2][0] = wmma_f16(a0, f2.v, acc[2][0]);
        acc[2][1] = wmma_f16(a1, f2.v, acc[2][1]);
        acc[3][0] = wmma_f16(a0, f3.v, acc[3][0]);
        acc[3][1] = wmma_f16(a1, f3.v, acc[3][1]);
    }
#pragma unroll
    for (int t = 0; t < 4; ++t)
#pragma unroll
        for (int m = 0; m < 2; ++m)
#pragma unroll
            for (int r = 0; r < 8; ++r)
                logits[(size_t)(Mbase + m * 16 + r + 8 * khalf) * kC + t * 16 + lrow] =
                    acc[t][m][r];
}

// One wave per row: logsumexp, entropy, gumbel argmax, one-hot, cond, logp accumulation.
__global__ __launch_bounds__(256) void sample_kernel(const float* __restrict__ logits,
                                                     const float* __restrict__ gum,
                                                     const float* __restrict__ pW,   // [P,C]
                                                     const float* __restrict__ pb,   // [P]
                                                     float* __restrict__ out_scalar,
                                                     float* __restrict__ out_oh,
                                                     float* __restrict__ out_ent,
                                                     float* __restrict__ out_tp,
                                                     float* __restrict__ cond,
                                                     float* __restrict__ lacc,
                                                     int step) {
    int wave = threadIdx.x >> 5, lane = threadIdx.x & 31;
    int row = blockIdx.x * 8 + wave;
    const float* lr = logits + (size_t)row * kC;
    float l0 = lr[lane], l1 = lr[lane + 32];
    float g0 = gum[(size_t)row * kC + lane], g1 = gum[(size_t)row * kC + lane + 32];

    // logsumexp
    float m = fmaxf(l0, l1);
#pragma unroll
    for (int d = 16; d; d >>= 1) m = fmaxf(m, __shfl_xor(m, d, 32));
    float se = __expf(l0 - m) + __expf(l1 - m);
#pragma unroll
    for (int d = 16; d; d >>= 1) se += __shfl_xor(se, d, 32);
    float lse = m + __logf(se);
    // entropy = lse - sum(p * l)
    float pl = __expf(l0 - lse) * l0 + __expf(l1 - lse) * l1;
#pragma unroll
    for (int d = 16; d; d >>= 1) pl += __shfl_xor(pl, d, 32);
    float ent = lse - pl;

    // argmax of logits + gumbel (first-index tie-break)
    float zv; int zi;
    {
        float z0 = l0 + g0, z1 = l1 + g1;
        if (z0 >= z1) { zv = z0; zi = lane; } else { zv = z1; zi = lane + 32; }
#pragma unroll
        for (int d = 16; d; d >>= 1) {
            float ov = __shfl_xor(zv, d, 32);
            int   oi = __shfl_xor(zi, d, 32);
            if (ov > zv || (ov == zv && oi < zi)) { zv = ov; zi = oi; }
        }
    }
    int amax = zi;
    float cand = (amax < 32) ? l0 : l1;          // amax uniform across wave
    float lmax = __shfl(cand, amax & 31, 32);
    float lp = lmax - lse;

    out_oh[(size_t)row * kC + lane]      = (lane == amax)      ? 1.0f : 0.0f;
    out_oh[(size_t)row * kC + lane + 32] = (lane + 32 == amax) ? 1.0f : 0.0f;
    if (lane == 0) {
        out_scalar[row] = (float)amax;
        out_ent[row] = ent;
        float acc = (step == 0) ? lp : (lacc[row] + lp);
        lacc[row] = acc;
        if (step == kS - 1) out_tp[row] = __expf(acc);
    }
    if (lane < kP) cond[(size_t)row * kP + lane] = pW[lane * kC + amax] + pb[lane];
}

// ---------------- host launch ----------------
extern "C" void kernel_launch(void* const* d_in, const int* in_sizes, int n_in,
                              void* d_out, int out_size, void* d_ws, size_t ws_size,
                              hipStream_t stream) {
    const float* t     = (const float*)d_in[0];
    const float* gum   = (const float*)d_in[1];
    const float* teW1  = (const float*)d_in[2];
    const float* teB1  = (const float*)d_in[3];
    const float* teW2  = (const float*)d_in[4];
    const float* teB2  = (const float*)d_in[5];
    const float* W_ih  = (const float*)d_in[6];
    const float* W_hh  = (const float*)d_in[7];
    const float* b_ih  = (const float*)d_in[8];
    const float* b_hh  = (const float*)d_in[9];
    const float* projW = (const float*)d_in[10];
    const float* projB = (const float*)d_in[11];
    const float* so1W  = (const float*)d_in[12];
    const float* so1b  = (const float*)d_in[13];
    const float* so2W  = (const float*)d_in[14];
    const float* so2b  = (const float*)d_in[15];
    float* out = (float*)d_out;

    char* ws = (char*)d_ws;
    size_t off = 0;
    auto alloc = [&](size_t bytes) -> void* {
        off = (off + 255) & ~(size_t)255;
        void* p = ws + off;
        off += bytes;
        return p;
    };
    _Float16* whh16  = (_Float16*)alloc((size_t)4 * kH * kH * 2);
    _Float16* tew216 = (_Float16*)alloc((size_t)kH * kH * 2);
    float*    pe     = (float*)alloc((size_t)kB * kP * 4);
    float*    cond   = (float*)alloc((size_t)kB * kP * 4);
    _Float16* x1     = (_Float16*)alloc((size_t)kB * kH * 2);
    _Float16* hbuf0  = (_Float16*)alloc((size_t)kB * kH * 2);
    _Float16* hbuf1  = (_Float16*)alloc((size_t)kB * kH * 2);
    float*    cbuf   = (float*)alloc((size_t)kB * kH * 4);
    _Float16* so16   = (_Float16*)alloc((size_t)kC * kH * 2);
    float*    logits = (float*)alloc((size_t)kB * kC * 4);
    float*    lacc   = (float*)alloc((size_t)kB * 4);
    _Float16* hb[2] = {hbuf0, hbuf1};

    cvt_f32_to_f16<<<(4 * kH * kH + 255) / 256, 256, 0, stream>>>(W_hh, whh16, 4 * kH * kH);
    cvt_f32_to_f16<<<(kH * kH + 255) / 256, 256, 0, stream>>>(teW2, tew216, kH * kH);
    posenc_kernel<<<(kB + 255) / 256, 256, 0, stream>>>(t, pe);
    zero_f32<<<(kB * kP + 255) / 256, 256, 0, stream>>>(cond, kB * kP);
    te1_kernel<<<(kB * kH) / 256, 256, 0, stream>>>(pe, teW1, teB1, x1);
    te2_kernel<<<8192, 256, 0, stream>>>(x1, tew216, teB2, cbuf, hb[0]);

    float* out_scalar = out;
    float* out_oh     = out + (size_t)kS * kB;
    float* out_ent    = out + (size_t)kS * kB + (size_t)kS * kB * kC;
    float* out_tp     = out + (size_t)kS * kB + (size_t)kS * kB * kC + (size_t)kS * kB;

    for (int s = 0; s < kS; ++s) {
        species_kernel<<<kC, 256, 0, stream>>>(so1W + (size_t)s * kH, so1b + (size_t)s * kH,
                                               so2W + (size_t)s * kH * kH, so2b + (size_t)s * kH,
                                               so16);
        lstm_kernel<<<4096, 256, 0, stream>>>(hb[s & 1], whh16, pe, cond,
                                              W_ih, b_ih, b_hh, cbuf, hb[(s + 1) & 1]);
        logits_kernel<<<128, 256, 0, stream>>>(hb[(s + 1) & 1], so16, logits);
        sample_kernel<<<kB / 8, 256, 0, stream>>>(logits, gum + (size_t)s * kB * kC,
                                                  projW + (size_t)s * kP * kC,
                                                  projB + (size_t)s * kP,
                                                  out_scalar + (size_t)s * kB,
                                                  out_oh + (size_t)s * kB * kC,
                                                  out_ent + (size_t)s * kB,
                                                  out_tp, cond, lacc, s);
    }
}
